// HSMSSD_71992241816068
// MI455X (gfx1250) — matrix operand order repaired and checked by reading, not compiled, not run on hardware
//
#include <hip/hip_runtime.h>
#include <hip/hip_bf16.h>

typedef __bf16 bf16_t;
typedef __attribute__((ext_vector_type(16))) __bf16 v16bf;
typedef __attribute__((ext_vector_type(8)))  __bf16 v8bf;
typedef __attribute__((ext_vector_type(8)))  float  v8f;

#define BB  32
#define DMD 256
#define SSD 64
#define LLD 4096
#define C3S 192

// round-half-up f32 -> bf16, packed pair in one dword (1 v_perm + 2 v_add)
__device__ __forceinline__ unsigned pk2(float a, float b) {
  union { float f; unsigned u; } ua, ub; ua.f = a; ub.f = b;
  return __builtin_amdgcn_perm(ub.u + 0x8000u, ua.u + 0x8000u, 0x07060302u);
}
__device__ __forceinline__ uint4 pk8(float f0,float f1,float f2,float f3,
                                     float f4,float f5,float f6,float f7) {
  uint4 u; u.x = pk2(f0,f1); u.y = pk2(f2,f3); u.z = pk2(f4,f5); u.w = pk2(f6,f7);
  return u;
}
// RNE scalar (used in elementwise kernels and the k_gemm_out epilogue)
__device__ __forceinline__ bf16_t tobf(float x) {
  union { float f; unsigned u; } v; v.f = x;
  unsigned r = v.u + 0x7FFFu + ((v.u >> 16) & 1u);
  unsigned short hs = (unsigned short)(r >> 16);
  bf16_t out; __builtin_memcpy(&out, &hs, 2);
  return out;
}
// 32B fragment load from LDS (2x ds_load_b128)
__device__ __forceinline__ v16bf ld_frag(const bf16_t* p) {
  v8bf lo = *(const v8bf*)(p);
  v8bf hi = *(const v8bf*)(p + 8);
  return __builtin_shufflevector(lo, hi, 0,1,2,3,4,5,6,7,8,9,10,11,12,13,14,15);
}
#define SWAP2(c) ((((c) & 1) << 1) | ((c) >> 1))

// ---------------------------------------------------------------------------
// Kernel 1: BCdt[b,o,l] = sum_d W[o,d]*x[b,d,l] + bias[o]
// M=192(3x64), N=4096(32x128), K=256. 256 thr = 8 waves (4M x 2N).
// ---------------------------------------------------------------------------
__global__ void __launch_bounds__(256) k_gemm_bcdt(
    const float* __restrict__ x, const float* __restrict__ Wt,
    const float* __restrict__ bias, float* __restrict__ out)
{
  __shared__ __align__(16) bf16_t As[64][40];
  __shared__ __align__(16) bf16_t Bsn[128][40];
  int idx = blockIdx.x;
  int nt = idx & 31, mt = (idx >> 5) % 3, b = idx / 96;
  int t = threadIdx.x, lane = t & 31, wave = t >> 5;
  int wm = wave >> 1, wn = wave & 1;
  int g = lane >> 4, ln = lane & 15;
  int m0 = mt * 64, n0 = nt * 128;

  v8f acc[4];
  for (int i = 0; i < 4; ++i)
    for (int e = 0; e < 8; ++e) acc[i][e] = 0.f;

  for (int kb = 0; kb < DMD; kb += 32) {
    { // stage A: one 8-chunk per thread
      int m = t >> 2, kc = t & 3, pc = kc;
      const float4* gp = (const float4*)&Wt[(m0 + m) * DMD + kb + kc * 8];
      float4 a0 = gp[0], a1 = gp[1];
      *(uint4*)&As[m][pc * 8] = pk8(a0.x,a0.y,a0.z,a0.w,a1.x,a1.y,a1.z,a1.w);
    }
    for (int c = t; c < 512; c += 256) { // stage B transposed
      int n = c & 127, kc = c >> 7;
      const float* gp = &x[((size_t)(b * DMD + kb + kc * 8)) * LLD + n0 + n];
      *(uint4*)&Bsn[n][kc * 8] =
          pk8(gp[0], gp[LLD], gp[2*LLD], gp[3*LLD],
              gp[4*LLD], gp[5*LLD], gp[6*LLD], gp[7*LLD]);
    }
    __syncthreads();
    v16bf af = ld_frag(&As[wm * 16 + ln][16 * g]);
    #pragma unroll
    for (int ntl = 0; ntl < 4; ++ntl) {
      v16bf bfr = ld_frag(&Bsn[wn * 64 + ntl * 16 + ln][16 * g]);
      acc[ntl] = __builtin_amdgcn_wmma_f32_16x16x32_bf16(
          false, af, false, bfr, (short)0, acc[ntl], false, false);
    }
    __syncthreads();
  }
  #pragma unroll
  for (int ntl = 0; ntl < 4; ++ntl)
    #pragma unroll
    for (int r = 0; r < 8; ++r) {
      int o = m0 + wm * 16 + 8 * g + r;
      int l = n0 + wn * 64 + ntl * 16 + ln;
      out[((size_t)(b * C3S + o)) * LLD + l] = acc[ntl][r] + bias[o];
    }
}

// ---------------------------------------------------------------------------
// Kernel 2: depthwise 3x3 conv, pad 1, on [B,192,64,64]
// ---------------------------------------------------------------------------
__global__ void __launch_bounds__(256) k_dwconv(
    const float* __restrict__ in, const float* __restrict__ w,
    const float* __restrict__ bias, float* __restrict__ out)
{
  int idx = blockIdx.x * 256 + threadIdx.x;
  int xw = idx & 63, yh = (idx >> 6) & 63;
  int c = (idx >> 12) % C3S, b = (idx >> 12) / C3S;
  const float* ip = in + ((size_t)(b * C3S + c)) * 4096;
  const float* wp = w + c * 9;
  float acc = bias[c];
  #pragma unroll
  for (int dy = -1; dy <= 1; ++dy) {
    int yy = yh + dy;
    if (yy < 0 || yy > 63) continue;
    #pragma unroll
    for (int dx = -1; dx <= 1; ++dx) {
      int xx = xw + dx;
      if (xx < 0 || xx > 63) continue;
      acc += wp[(dy + 1) * 3 + (dx + 1)] * ip[yy * 64 + xx];
    }
  }
  out[idx] = acc;
}

// ---------------------------------------------------------------------------
// Kernel 3: AB[b,s,l] = softmax_l(dt+A[s]) * Bm  -> bf16 (k contiguous)
// ---------------------------------------------------------------------------
__global__ void __launch_bounds__(256) k_softmax_ab(
    const float* __restrict__ bcdt2, const float* __restrict__ A,
    bf16_t* __restrict__ ab)
{
  __shared__ float red[256];
  int row = blockIdx.x;                 // b*64 + s
  int b = row >> 6, s = row & 63;
  int t = threadIdx.x;
  const float* dt = bcdt2 + ((size_t)(b * C3S + 128 + s)) * LLD;
  const float* Bm = bcdt2 + ((size_t)(b * C3S + s)) * LLD;
  float a = A[s];
  float loc[16];
  float mx = -3.4e38f;
  #pragma unroll
  for (int i = 0; i < 16; ++i) {
    float v = dt[i * 256 + t] + a;
    loc[i] = v; mx = fmaxf(mx, v);
  }
  red[t] = mx; __syncthreads();
  for (int st = 128; st > 0; st >>= 1) {
    if (t < st) red[t] = fmaxf(red[t], red[t + st]);
    __syncthreads();
  }
  mx = red[0]; __syncthreads();
  float sum = 0.f;
  #pragma unroll
  for (int i = 0; i < 16; ++i) { float e = __expf(loc[i] - mx); loc[i] = e; sum += e; }
  red[t] = sum; __syncthreads();
  for (int st = 128; st > 0; st >>= 1) {
    if (t < st) red[t] += red[t + st];
    __syncthreads();
  }
  float inv = 1.0f / red[0];
  bf16_t* op = ab + ((size_t)row) * LLD;
  #pragma unroll
  for (int i = 0; i < 16; ++i)
    op[i * 256 + t] = tobf(loc[i] * inv * Bm[i * 256 + t]);
}

// ---------------------------------------------------------------------------
// Kernel 4: split-K partials of h[b,d,s] = sum_l x[b,d,l]*AB[b,s,l]
// grid = B*4*8 (mt of 64 rows, ks of 512 K).  128 thr = 4 waves.
// ---------------------------------------------------------------------------
__global__ void __launch_bounds__(128) k_gemm_h(
    const float* __restrict__ x, const bf16_t* __restrict__ ab,
    float* __restrict__ part)
{
  __shared__ __align__(16) bf16_t As[64][40];
  __shared__ __align__(16) bf16_t Bsn[64][40];
  int idx = blockIdx.x;
  int ks = idx & 7, mt = (idx >> 3) & 3, b = idx >> 5;
  int t = threadIdx.x, lane = t & 31, wave = t >> 5;
  int g = lane >> 4, ln = lane & 15;
  int m0 = mt * 64, kbeg = ks * 512;

  v8f acc[4];
  for (int i = 0; i < 4; ++i)
    for (int e = 0; e < 8; ++e) acc[i][e] = 0.f;

  for (int kb = kbeg; kb < kbeg + 512; kb += 32) {
    for (int c = t; c < 256; c += 128) {     // stage A (x)
      int m = c >> 2, kc = c & 3, pc = kc;
      const float4* gp = (const float4*)&x[((size_t)(b * DMD + m0 + m)) * LLD + kb + kc * 8];
      float4 a0 = gp[0], a1 = gp[1];
      *(uint4*)&As[m][pc * 8] = pk8(a0.x,a0.y,a0.z,a0.w,a1.x,a1.y,a1.z,a1.w);
    }
    for (int c = t; c < 256; c += 128) {     // stage B (ab copy, k contiguous)
      int n = c >> 2, kc = c & 3;
      *(uint4*)&Bsn[n][kc * 8] =
          *(const uint4*)&ab[((size_t)(b * SSD + n)) * LLD + kb + kc * 8];
    }
    __syncthreads();
    v16bf af = ld_frag(&As[wave * 16 + ln][16 * g]);
    #pragma unroll
    for (int ntl = 0; ntl < 4; ++ntl) {
      v16bf bfr = ld_frag(&Bsn[ntl * 16 + ln][16 * g]);
      acc[ntl] = __builtin_amdgcn_wmma_f32_16x16x32_bf16(
          false, af, false, bfr, (short)0, acc[ntl], false, false);
    }
    __syncthreads();
  }
  float* pp = part + (size_t)ks * (BB * DMD * SSD);
  #pragma unroll
  for (int ntl = 0; ntl < 4; ++ntl)
    #pragma unroll
    for (int r = 0; r < 8; ++r) {
      int d = m0 + wave * 16 + 8 * g + r, s = ntl * 16 + ln;
      pp[((size_t)(b * DMD + d)) * SSD + s] = acc[ntl][r];
    }
}

// Reduce split-K partials; write h (f32, d_out slice) + transposed bf16 copy
__global__ void __launch_bounds__(256) k_reduce_h(
    const float* __restrict__ part, float* __restrict__ hout,
    bf16_t* __restrict__ htr)
{
  int idx = blockIdx.x * 256 + threadIdx.x;   // (b*256+d)*64+s
  float s = 0.f;
  #pragma unroll
  for (int ks = 0; ks < 8; ++ks) s += part[(size_t)ks * (BB * DMD * SSD) + idx];
  hout[idx] = s;
  int si = idx & 63, d = (idx >> 6) & 255, b = idx >> 14;
  htr[((size_t)(b * SSD + si)) * DMD + d] = tobf(s);
}

// ---------------------------------------------------------------------------
// Kernel 5: hz[b,o,s] = sum_d hz_w[o,d]*htr[b,s,d] + hz_b[o]. M=512,N=64,K=256
// ---------------------------------------------------------------------------
__global__ void __launch_bounds__(128) k_gemm_hz(
    const bf16_t* __restrict__ htr, const float* __restrict__ w,
    const float* __restrict__ bias, float* __restrict__ out)
{
  __shared__ __align__(16) bf16_t As[64][40];
  __shared__ __align__(16) bf16_t Bsn[64][40];
  int b = blockIdx.x >> 3, mt = blockIdx.x & 7;
  int t = threadIdx.x, lane = t & 31, wave = t >> 5;
  int g = lane >> 4, ln = lane & 15;
  int m0 = mt * 64;

  v8f acc[4];
  for (int i = 0; i < 4; ++i)
    for (int e = 0; e < 8; ++e) acc[i][e] = 0.f;

  for (int kb = 0; kb < DMD; kb += 32) {
    for (int c = t; c < 256; c += 128) {     // stage A (hz_w)
      int m = c >> 2, kc = c & 3, pc = kc;
      const float4* gp = (const float4*)&w[(m0 + m) * DMD + kb + kc * 8];
      float4 a0 = gp[0], a1 = gp[1];
      *(uint4*)&As[m][pc * 8] = pk8(a0.x,a0.y,a0.z,a0.w,a1.x,a1.y,a1.z,a1.w);
    }
    for (int c = t; c < 256; c += 128) {     // stage B (htr copy)
      int n = c >> 2, kc = c & 3;
      *(uint4*)&Bsn[n][kc * 8] =
          *(const uint4*)&htr[((size_t)(b * SSD + n)) * DMD + kb + kc * 8];
    }
    __syncthreads();
    v16bf af = ld_frag(&As[wave * 16 + ln][16 * g]);
    #pragma unroll
    for (int ntl = 0; ntl < 4; ++ntl) {
      v16bf bfr = ld_frag(&Bsn[ntl * 16 + ln][16 * g]);
      acc[ntl] = __builtin_amdgcn_wmma_f32_16x16x32_bf16(
          false, af, false, bfr, (short)0, acc[ntl], false, false);
    }
    __syncthreads();
  }
  #pragma unroll
  for (int ntl = 0; ntl < 4; ++ntl)
    #pragma unroll
    for (int r = 0; r < 8; ++r) {
      int o = m0 + wave * 16 + 8 * g + r, s = ntl * 16 + ln;
      out[((size_t)(b * 512 + o)) * SSD + s] = acc[ntl][r] + bias[o];
    }
}

// ---------------------------------------------------------------------------
// Kernel 6: g = hp * silu(z), written transposed [b][s][o] in bf16
// ---------------------------------------------------------------------------
__global__ void __launch_bounds__(256) k_gate(
    const float* __restrict__ hz, bf16_t* __restrict__ gtr)
{
  int idx = blockIdx.x * 256 + threadIdx.x;   // (b*64+s)*256+o
  int o = idx & 255, s = (idx >> 8) & 63, b = idx >> 14;
  float hp = hz[((size_t)(b * 512 + o)) * 64 + s];
  float z  = hz[((size_t)(b * 512 + o + 256)) * 64 + s];
  float sig = 1.f / (1.f + __expf(-z));
  gtr[idx] = tobf(hp * z * sig);
}

// ---------------------------------------------------------------------------
// Kernel 7: ho[b,o,s] = sum_d out_w[o,d]*gtr[b,s,d] + out_b[o] -> bf16 [b][o][s]
// ---------------------------------------------------------------------------
__global__ void __launch_bounds__(128) k_gemm_out(
    const bf16_t* __restrict__ gtr, const float* __restrict__ w,
    const float* __restrict__ bias, bf16_t* __restrict__ out)
{
  __shared__ __align__(16) bf16_t As[64][40];
  __shared__ __align__(16) bf16_t Bsn[64][40];
  int b = blockIdx.x >> 2, mt = blockIdx.x & 3;
  int t = threadIdx.x, lane = t & 31, wave = t >> 5;
  int g = lane >> 4, ln = lane & 15;
  int m0 = mt * 64;

  v8f acc[4];
  for (int i = 0; i < 4; ++i)
    for (int e = 0; e < 8; ++e) acc[i][e] = 0.f;

  for (int kb = 0; kb < DMD; kb += 32) {
    for (int c = t; c < 256; c += 128) {
      int m = c >> 2, kc = c & 3, pc = kc;
      const float4* gp = (const float4*)&w[(m0 + m) * DMD + kb + kc * 8];
      float4 a0 = gp[0], a1 = gp[1];
      *(uint4*)&As[m][pc * 8] = pk8(a0.x,a0.y,a0.z,a0.w,a1.x,a1.y,a1.z,a1.w);
    }
    for (int c = t; c < 256; c += 128) {
      int n = c >> 2, kc = c & 3;
      *(uint4*)&Bsn[n][kc * 8] =
          *(const uint4*)&gtr[((size_t)(b * SSD + n)) * DMD + kb + kc * 8];
    }
    __syncthreads();
    v16bf af = ld_frag(&As[wave * 16 + ln][16 * g]);
    #pragma unroll
    for (int ntl = 0; ntl < 4; ++ntl) {
      v16bf bfr = ld_frag(&Bsn[ntl * 16 + ln][16 * g]);
      acc[ntl] = __builtin_amdgcn_wmma_f32_16x16x32_bf16(
          false, af, false, bfr, (short)0, acc[ntl], false, false);
    }
    __syncthreads();
  }
  #pragma unroll
  for (int ntl = 0; ntl < 4; ++ntl)
    #pragma unroll
    for (int r = 0; r < 8; ++r) {
      int o = m0 + wave * 16 + 8 * g + r, s = ntl * 16 + ln;
      out[((size_t)(b * DMD + o)) * SSD + s] = tobf(acc[ntl][r] + bias[o]);
    }
}

// ---------------------------------------------------------------------------
// Kernel 8: y[b,d,l] = sum_s ho[b,d,s]*Cm[b,s,l] + x[b,d,l]*D[d]
// M=256(4x64), N=4096(32x128), K=64. 256 thr = 8 waves (4M x 2N).
// ---------------------------------------------------------------------------
__global__ void __launch_bounds__(256) k_gemm_y(
    const bf16_t* __restrict__ ho, const float* __restrict__ bcdt2,
    const float* __restrict__ x, const float* __restrict__ Dv,
    float* __restrict__ y)
{
  __shared__ __align__(16) bf16_t Ah[64][72];
  __shared__ __align__(16) bf16_t Bsn[128][72];
  int idx = blockIdx.x;
  int nt = idx & 31, mt = (idx >> 5) & 3, b = idx >> 7;
  int t = threadIdx.x, lane = t & 31, wave = t >> 5;
  int wm = wave >> 1, wn = wave & 1;
  int g = lane >> 4, ln = lane & 15;
  int m0 = mt * 64, n0 = nt * 128;

  for (int c = t; c < 512; c += 256) {     // stage A (ho bf16 copy)
    int m = c >> 3, kc = c & 7;
    int pc = kc;
    *(uint4*)&Ah[m][pc * 8] =
        *(const uint4*)&ho[((size_t)(b * DMD + m0 + m)) * SSD + kc * 8];
  }
  for (int c = t; c < 1024; c += 256) {    // stage B (Cm transposed, f32->bf16)
    int n = c & 127, kc = c >> 7;
    const float* gp = &bcdt2[((size_t)(b * C3S + 64 + kc * 8)) * LLD + n0 + n];
    *(uint4*)&Bsn[n][kc * 8] =
        pk8(gp[0], gp[LLD], gp[2*LLD], gp[3*LLD],
            gp[4*LLD], gp[5*LLD], gp[6*LLD], gp[7*LLD]);
  }
  __syncthreads();

  v8f acc[4];
  for (int i = 0; i < 4; ++i)
    for (int e = 0; e < 8; ++e) acc[i][e] = 0.f;

  #pragma unroll
  for (int kb = 0; kb < 64; kb += 32) {
    v16bf af = ld_frag(&Ah[wm * 16 + ln][kb + 16 * g]);
    #pragma unroll
    for (int ntl = 0; ntl < 4; ++ntl) {
      v16bf bfr = ld_frag(&Bsn[wn * 64 + ntl * 16 + ln][kb + 16 * g]);
      acc[ntl] = __builtin_amdgcn_wmma_f32_16x16x32_bf16(
          false, af, false, bfr, (short)0, acc[ntl], false, false);
    }
  }
  #pragma unroll
  for (int ntl = 0; ntl < 4; ++ntl)
    #pragma unroll
    for (int r = 0; r < 8; ++r) {
      int d = m0 + wm * 16 + 8 * g + r;
      int l = n0 + wn * 64 + ntl * 16 + ln;
      size_t ix = ((size_t)(b * DMD + d)) * LLD + l;
      y[ix] = acc[ntl][r] + x[ix] * Dv[d];
    }
}

// ---------------------------------------------------------------------------
extern "C" void kernel_launch(void* const* d_in, const int* in_sizes, int n_in,
                              void* d_out, int out_size, void* d_ws, size_t ws_size,
                              hipStream_t stream) {
  const float* x      = (const float*)d_in[0];
  const float* BCdt_w = (const float*)d_in[1];
  const float* BCdt_b = (const float*)d_in[2];
  const float* dw_w   = (const float*)d_in[3];
  const float* dw_b   = (const float*)d_in[4];
  const float* hz_w   = (const float*)d_in[5];
  const float* hz_b   = (const float*)d_in[6];
  const float* out_w  = (const float*)d_in[7];
  const float* out_b  = (const float*)d_in[8];
  const float* Av     = (const float*)d_in[9];
  const float* Dv     = (const float*)d_in[10];

  char* ws = (char*)d_ws;
  float*  bcdt  = (float*)(ws);                  // [32,192,4096] f32    96 MiB
  float*  bcdt2 = (float*)(ws + 100663296);      // [32,192,4096] f32    96 MiB
  bf16_t* ab    = (bf16_t*)(ws + 201326592);     // [32,64,4096]  bf16   16 MiB
  float*  part  = (float*)(ws + 218103808);      // [8][32,256,64] f32   16 MiB
  float*  hzb   = (float*)(ws + 234881024);      // [32,512,64]   f32     4 MiB
  bf16_t* htr   = (bf16_t*)(ws + 239075328);     // [32,64,256]   bf16    1 MiB
  bf16_t* gtr   = (bf16_t*)(ws + 240123904);     // [32,64,256]   bf16    1 MiB
  bf16_t* hob   = (bf16_t*)(ws + 241172480);     // [32,256,64]   bf16    1 MiB

  float* yout = (float*)d_out;                   // [32,256,4096]
  float* hout = yout + (size_t)BB * DMD * LLD;   // [32,256,64]

  k_gemm_bcdt <<<dim3(BB * 3 * 32),          dim3(256), 0, stream>>>(x, BCdt_w, BCdt_b, bcdt);
  k_dwconv    <<<dim3(BB * C3S * LLD / 256), dim3(256), 0, stream>>>(bcdt, dw_w, dw_b, bcdt2);
  k_softmax_ab<<<dim3(BB * SSD),             dim3(256), 0, stream>>>(bcdt2, Av, ab);
  k_gemm_h    <<<dim3(BB * 4 * 8),           dim3(128), 0, stream>>>(x, ab, part);
  k_reduce_h  <<<dim3(BB * DMD * SSD / 256), dim3(256), 0, stream>>>(part, hout, htr);
  k_gemm_hz   <<<dim3(BB * 8),               dim3(128), 0, stream>>>(htr, hz_w, hz_b, hzb);
  k_gate      <<<dim3(BB * SSD * DMD / 256), dim3(256), 0, stream>>>(hzb, gtr);
  k_gemm_out  <<<dim3(BB * 4),               dim3(128), 0, stream>>>(gtr, out_w, out_b, hob);
  k_gemm_y    <<<dim3(BB * 4 * 32),          dim3(256), 0, stream>>>(hob, bcdt2, x, Dv, yout);
}
